// MultiHeadAttentionWithKVCache_30932354465932
// MI455X (gfx1250) — compile-verified, matrix operand order checked by reading, not run on hardware
//
#include <hip/hip_runtime.h>

typedef __attribute__((ext_vector_type(16))) __bf16 v16bf;
typedef __attribute__((ext_vector_type(8)))  __bf16 v8bf;
typedef __attribute__((ext_vector_type(8)))  float  v8f;

#define DD     4096
#define BB     8
#define TT     8
#define HH     32
#define HD     128
#define TCC    4096
#define STOT   4104
#define BT     64
#define NCHUNK 129   // ceil(4104/32)

#define CAT16(lo, hi) __builtin_shufflevector((lo), (hi), 0, 1, 2, 3, 4, 5, 6, 7, \
                                              8, 9, 10, 11, 12, 13, 14, 15)

// C = A[64,4096] @ W[4096,4096] + bias
// MODE 0: row-major [64,4096] out ; MODE 1: scatter into KV cache tail
template <int MODE>
__global__ __launch_bounds__(32)
void proj_gemm(const float* __restrict__ A, const float* __restrict__ W,
               const float* __restrict__ bias, float* __restrict__ outp) {
  const int n0   = blockIdx.x * 16;
  const int m0   = blockIdx.y * 16;
  const int lane = threadIdx.x;
  const int half = lane >> 4;
  const int l16  = lane & 15;

  __shared__ alignas(16) __bf16 lA[16 * 32];    // [m][k]
  __shared__ alignas(16) __bf16 lBt[16 * 32];   // [n][k]  (W tile transposed)

  const int arow = lane >> 1;          // 0..15
  const int acol = (lane & 1) * 16;    // 0 or 16

  v8f acc = {};
  for (int k0 = 0; k0 < DD; k0 += 32) {
    // ---- stage A: each lane 16 consecutive floats -> two b128 LDS stores
    float fa[16];
    const float* ap = A + (size_t)(m0 + arow) * DD + k0 + acol;
    *(float4*)(&fa[0])  = *(const float4*)(ap + 0);
    *(float4*)(&fa[4])  = *(const float4*)(ap + 4);
    *(float4*)(&fa[8])  = *(const float4*)(ap + 8);
    *(float4*)(&fa[12]) = *(const float4*)(ap + 12);
    v8bf h0, h1;
#pragma unroll
    for (int j = 0; j < 8; ++j) { h0[j] = (__bf16)fa[j]; h1[j] = (__bf16)fa[j + 8]; }
    *(v8bf*)(&lA[arow * 32 + acol])     = h0;
    *(v8bf*)(&lA[arow * 32 + acol + 8]) = h1;

    // ---- stage W transposed: lane reads row k0+lane (16 floats), scatters to [n][k]
    float fw[16];
    const float* wp = W + (size_t)(k0 + lane) * DD + n0;
    *(float4*)(&fw[0])  = *(const float4*)(wp + 0);
    *(float4*)(&fw[4])  = *(const float4*)(wp + 4);
    *(float4*)(&fw[8])  = *(const float4*)(wp + 8);
    *(float4*)(&fw[12]) = *(const float4*)(wp + 12);
#pragma unroll
    for (int j = 0; j < 16; ++j) lBt[j * 32 + lane] = (__bf16)fw[j];
    __syncthreads();

    // ---- WMMA operands: two aligned b128 LDS loads each, register concat
    v8bf alo = *(const v8bf*)(&lA[l16 * 32 + half * 8]);
    v8bf ahi = *(const v8bf*)(&lA[l16 * 32 + 16 + half * 8]);
    v8bf blo = *(const v8bf*)(&lBt[l16 * 32 + half * 8]);
    v8bf bhi = *(const v8bf*)(&lBt[l16 * 32 + 16 + half * 8]);
    v16bf av = CAT16(alo, ahi);
    v16bf bv = CAT16(blo, bhi);
    acc = __builtin_amdgcn_wmma_f32_16x16x32_bf16(false, av, false, bv,
                                                  (short)0, acc, false, false);
    __syncthreads();
  }

  const float bz = bias[n0 + l16];
#pragma unroll
  for (int r = 0; r < 8; ++r) {
    float v = acc[r] + bz;
    int gm = m0 + half * 8 + r;        // 0..63
    int gn = n0 + l16;                 // 0..4095
    if (MODE == 0) {
      outp[gm * DD + gn] = v;
    } else {
      int b = gm >> 3, t = gm & 7;
      int h = gn >> 7, d = gn & 127;
      outp[(((size_t)(b * HH + h) * STOT) + TCC + t) * HD + d] = v;
    }
  }
}

// One block per (b,h). Fuses past_k/past_v -> out-cache copy with flash
// attention (bf16 WMMA for QK^T and PV, online softmax, cross-wave combine).
__global__ __launch_bounds__(256)
void attn_fused(const float* __restrict__ past_k, const float* __restrict__ past_v,
                const float* __restrict__ q, float* __restrict__ kc,
                float* __restrict__ vc, float* __restrict__ ctx) {
  const int bh   = blockIdx.x;
  const int b    = bh >> 5, h = bh & 31;
  const int lane = threadIdx.x & 31;
  const int wave = threadIdx.x >> 5;
  const int half = lane >> 4, l16 = lane & 15;

  extern __shared__ char smem[];
  char*   wbase = smem + wave * (18 * 1024);
  __bf16* lK  = (__bf16*)wbase;                  // [s=32][d=128] bf16 (K tile)
  __bf16* lVt = (__bf16*)(wbase + 8 * 1024);     // [d=128][s=32] bf16 (V transposed)
  __bf16* lPb = (__bf16*)(wbase + 16 * 1024);    // [m=16][s=32] bf16 (P tile)
  float*  lM  = (float*)(smem + 8 * 18 * 1024);  // [8 waves][16 rows]
  float*  lL  = lM + 8 * 16;

  const float* pk = past_k + (size_t)bh * TCC * HD;
  const float* pv = past_v + (size_t)bh * TCC * HD;
  float* ok = kc + (size_t)bh * STOT * HD;
  float* ov = vc + (size_t)bh * STOT * HD;

  // q as WMMA A tiles, pre-scaled by 1/sqrt(HD); rows >= T zero-padded
  const float scale = 0.088388347648318447f;
  v16bf a[4];
#pragma unroll
  for (int dc = 0; dc < 4; ++dc) {
#pragma unroll
    for (int e = 0; e < 16; ++e) {
      int kk = (e < 8) ? (half * 8 + e) : (8 + half * 8 + e);
      int d  = dc * 32 + kk;
      float qv = (l16 < TT) ? q[((b * TT + l16) * HH + h) * HD + d] * scale : 0.0f;
      a[dc][e] = (__bf16)qv;
    }
  }

  float rm[8], rl[8];
  v8f cacc[8];
#pragma unroll
  for (int r = 0; r < 8; ++r) { rm[r] = -1e30f; rl[r] = 0.0f; }
#pragma unroll
  for (int nc = 0; nc < 8; ++nc) cacc[nc] = (v8f){};

  for (int ch = wave; ch < NCHUNK; ch += 8) {
    const int s0   = ch * 32;
    const int srow = s0 + lane;                  // this lane's K/V row
    const bool valid = srow < STOT;
    const bool past  = srow < TCC;

    // ---- stream K/V: fused cache copy + bf16 LDS staging
#pragma unroll 2
    for (int j = 0; j < HD; j += 8) {
      float fk[8], fv[8];
      if (valid) {
        const size_t off = (size_t)srow * HD + j;
        if (past) {
          *(float4*)(&fk[0]) = *(const float4*)(pk + off);
          *(float4*)(&fk[4]) = *(const float4*)(pk + off + 4);
          *(float4*)(&fv[0]) = *(const float4*)(pv + off);
          *(float4*)(&fv[4]) = *(const float4*)(pv + off + 4);
          *(float4*)(ok + off)     = *(const float4*)(&fk[0]);   // fused copy
          *(float4*)(ok + off + 4) = *(const float4*)(&fk[4]);
          *(float4*)(ov + off)     = *(const float4*)(&fv[0]);
          *(float4*)(ov + off + 4) = *(const float4*)(&fv[4]);
        } else {                                   // new rows written by proj
          *(float4*)(&fk[0]) = *(const float4*)(ok + off);
          *(float4*)(&fk[4]) = *(const float4*)(ok + off + 4);
          *(float4*)(&fv[0]) = *(const float4*)(ov + off);
          *(float4*)(&fv[4]) = *(const float4*)(ov + off + 4);
        }
      } else {
#pragma unroll
        for (int c = 0; c < 8; ++c) { fk[c] = 0.f; fv[c] = 0.f; }
      }
      v8bf hk;
#pragma unroll
      for (int c = 0; c < 8; ++c) hk[c] = (__bf16)fk[c];
      *(v8bf*)(&lK[lane * HD + j]) = hk;                         // b128 store
#pragma unroll
      for (int c = 0; c < 8; ++c) lVt[(j + c) * 32 + lane] = (__bf16)fv[c];
    }
    asm volatile("s_wait_dscnt 0" ::: "memory");  // LDS stores visible to own wave

    // ---- S = q @ K^T : two 16-wide s tiles x four K-chunks of 32
    v8f sc[2]; sc[0] = (v8f){}; sc[1] = (v8f){};
#pragma unroll
    for (int tile = 0; tile < 2; ++tile) {
#pragma unroll
      for (int dc = 0; dc < 4; ++dc) {
        const int rowoff = (tile * 16 + l16) * HD + dc * 32 + half * 8;
        v8bf blo = *(const v8bf*)(&lK[rowoff]);
        v8bf bhi = *(const v8bf*)(&lK[rowoff + 16]);
        v16bf bv = CAT16(blo, bhi);
        sc[tile] = __builtin_amdgcn_wmma_f32_16x16x32_bf16(false, a[dc], false, bv,
                                                           (short)0, sc[tile], false, false);
      }
    }

    // ---- mask tail + online softmax (wave32 xor reductions over 16 lanes)
    const bool in0 = (s0 + l16) < STOT;
    const bool in1 = (s0 + 16 + l16) < STOT;
#pragma unroll
    for (int r = 0; r < 8; ++r) {
      if (!in0) sc[0][r] = -1e30f;
      if (!in1) sc[1][r] = -1e30f;
      float mx = fmaxf(sc[0][r], sc[1][r]);
#pragma unroll
      for (int off = 1; off < 16; off <<= 1)
        mx = fmaxf(mx, __shfl_xor(mx, off, 32));
      float M    = fmaxf(rm[r], mx);
      float corr = __expf(rm[r] - M);
      float p0   = __expf(sc[0][r] - M);
      float p1   = __expf(sc[1][r] - M);
      float ps   = p0 + p1;
#pragma unroll
      for (int off = 1; off < 16; off <<= 1)
        ps += __shfl_xor(ps, off, 32);
      rl[r] = rl[r] * corr + ps;
      rm[r] = M;
#pragma unroll
      for (int nc = 0; nc < 8; ++nc) cacc[nc][r] *= corr;
      lPb[(r + half * 8) * 32 + l16]      = (__bf16)p0;   // P as bf16, s innermost
      lPb[(r + half * 8) * 32 + 16 + l16] = (__bf16)p1;
    }
    asm volatile("s_wait_dscnt 0" ::: "memory");

    // ---- ctx += P @ V : P as A-matrix, 8 n-tiles of 16 over HD=128
    v8bf plo = *(const v8bf*)(&lPb[l16 * 32 + half * 8]);
    v8bf phi = *(const v8bf*)(&lPb[l16 * 32 + 16 + half * 8]);
    v16bf pa = CAT16(plo, phi);
#pragma unroll
    for (int nc = 0; nc < 8; ++nc) {
      const int coloff = (nc * 16 + l16) * 32 + half * 8;
      v8bf blo = *(const v8bf*)(&lVt[coloff]);
      v8bf bhi = *(const v8bf*)(&lVt[coloff + 16]);
      v16bf bv = CAT16(blo, bhi);
      cacc[nc] = __builtin_amdgcn_wmma_f32_16x16x32_bf16(false, pa, false, bv,
                                                         (short)0, cacc[nc], false, false);
    }
    asm volatile("s_wait_dscnt 0" ::: "memory");
  }

  // ---- per-wave partials to LDS (ctx aliases this wave's K staging region)
  float* lCtx = (float*)wbase;                   // 16x128 f32
#pragma unroll
  for (int nc = 0; nc < 8; ++nc)
#pragma unroll
    for (int r = 0; r < 8; ++r)
      lCtx[(r + half * 8) * HD + nc * 16 + l16] = cacc[nc][r];
  if (l16 == 0) {
#pragma unroll
    for (int r = 0; r < 8; ++r) {
      lM[wave * 16 + half * 8 + r] = rm[r];
      lL[wave * 16 + half * 8 + r] = rl[r];
    }
  }
  __syncthreads();

  // ---- cross-wave flash combine, valid rows t < 8 only
  for (int i = threadIdx.x; i < TT * HD; i += 256) {
    int t = i >> 7, d = i & 127;
    float M = -1e30f;
#pragma unroll
    for (int w = 0; w < 8; ++w) M = fmaxf(M, lM[w * 16 + t]);
    float Ls = 0.f, C = 0.f;
#pragma unroll
    for (int w = 0; w < 8; ++w) {
      float f = __expf(lM[w * 16 + t] - M);
      Ls += lL[w * 16 + t] * f;
      C  += ((const float*)(smem + w * 18 * 1024))[t * HD + d] * f;
    }
    ctx[((b * TT + t) * HH + h) * HD + d] = C / Ls;   // [B,T,H,HD] == [64,4096]
  }
}

extern "C" void kernel_launch(void* const* d_in, const int* in_sizes, int n_in,
                              void* d_out, int out_size, void* d_ws, size_t ws_size,
                              hipStream_t stream) {
  const float* x  = (const float*)d_in[0];
  const float* pk = (const float*)d_in[1];
  const float* pv = (const float*)d_in[2];
  const float* Wq = (const float*)d_in[3];
  const float* Wk = (const float*)d_in[4];
  const float* Wv = (const float*)d_in[5];
  const float* Wo = (const float*)d_in[6];
  const float* bq = (const float*)d_in[7];
  const float* bk = (const float*)d_in[8];
  const float* bv = (const float*)d_in[9];
  const float* bo = (const float*)d_in[10];

  float* out = (float*)d_out;                      // [B,T,D]
  const size_t KV_ELEMS = (size_t)BB * HH * STOT * HD;
  float* kc = out + (size_t)BB * TT * DD;          // k cache output
  float* vc = kc + KV_ELEMS;                       // v cache output

  float* q_ws   = (float*)d_ws;                    // [64,4096] as [B,T,H,HD]
  float* ctx_ws = q_ws + (size_t)BT * DD;          // [64,4096] as [B,T,H,HD]

  dim3 ggrid(DD / 16, BT / 16);
  proj_gemm<0><<<ggrid, 32, 0, stream>>>(x, Wq, bq, q_ws);
  proj_gemm<1><<<ggrid, 32, 0, stream>>>(x, Wk, bk, kc);   // k_new -> cache tail
  proj_gemm<1><<<ggrid, 32, 0, stream>>>(x, Wv, bv, vc);   // v_new -> cache tail

  size_t smem_bytes = 8 * 18 * 1024 + 2 * 8 * 16 * sizeof(float); // 145 KB
  attn_fused<<<BB * HH, 256, smem_bytes, stream>>>(pk, pv, q_ws, kc, vc, ctx_ws);

  proj_gemm<0><<<ggrid, 32, 0, stream>>>(ctx_ws, Wo, bo, out);
}
